// Attention_13606456393885
// MI455X (gfx1250) — matrix operand order repaired and checked by reading, not compiled, not run on hardware
//
#include <hip/hip_runtime.h>

// ---------------------------------------------------------------------------
// nGPT-style attention forward for gfx1250 (MI455X), bf16 WMMA pipeline.
// B=8, N=1024, DIM=768, HEADS=12, HEAD_DIM=64.
// ---------------------------------------------------------------------------

#define DIM    768
#define HEADS  12
#define HD     64
#define BATCH  8
#define SEQ    1024
#define MROWS  (BATCH * SEQ)          // 8192

typedef __attribute__((ext_vector_type(16))) __bf16 v16bf;
typedef __attribute__((ext_vector_type(8)))  float  v8f;

union FragB16 {
    uint4 u[2];   // 32 bytes
    v16bf v;      // 16 bf16
};

__device__ __forceinline__ unsigned short f2bf(float f) {
    unsigned int u = __float_as_uint(f);
    u += 0x7fffu + ((u >> 16) & 1u);          // round-to-nearest-even
    return (unsigned short)(u >> 16);
}

// ---------------------------------------------------------------------------
// f32 -> bf16 elementwise convert (grid-stride)
// ---------------------------------------------------------------------------
__global__ void __launch_bounds__(256)
cvt_f32_bf16_kernel(const float* __restrict__ src,
                    unsigned short* __restrict__ dst, int n) {
    int i = blockIdx.x * blockDim.x + threadIdx.x;
    int stride = gridDim.x * blockDim.x;
    for (; i < n; i += stride) dst[i] = f2bf(src[i]);
}

// ---------------------------------------------------------------------------
// C[M,N] f32 = A[M,K] bf16 (row-major) x W[N,K]^T (bf16 row-major, torch Linear)
// block = 256 threads = 8 waves; wave (wr,wc) owns rows [wr*16,+16), cols [wc*32,+32)
// of a 64x64 block tile. grid = (M/64, N/64).
// ---------------------------------------------------------------------------
__global__ void __launch_bounds__(256)
gemm_nt_bf16_kernel(const unsigned short* __restrict__ A,
                    const unsigned short* __restrict__ W,
                    float* __restrict__ C,
                    int M, int N, int K) {
    const int lane = threadIdx.x & 31;
    const int wave = threadIdx.x >> 5;
    const int wr   = wave & 3;
    const int wc   = wave >> 2;
    const int h    = lane >> 4;      // 16-lane half
    const int ln   = lane & 15;
    const int mBase = blockIdx.x * 64 + wr * 16;
    const int nBase = blockIdx.y * 64 + wc * 32;

    v8f c0 = {}; v8f c1 = {};

    const unsigned short* arow  = A + (size_t)(mBase + ln) * K;       // A lane row = M index
    const unsigned short* wrow0 = W + (size_t)(nBase + ln) * K;       // B lane col = W row
    const unsigned short* wrow1 = W + (size_t)(nBase + 16 + ln) * K;

    for (int kk = 0; kk < K; kk += 32) {
        FragB16 a, b0, b1;
        // A 16-bit 16x32 layout: elems 0..7 -> K = kk+8h+j ; elems 8..15 -> K = kk+8h+16+j
        a.u[0]  = *reinterpret_cast<const uint4*>(arow  + kk + 8 * h);
        a.u[1]  = *reinterpret_cast<const uint4*>(arow  + kk + 8 * h + 16);
        // B 16-bit 32x16 layout: elems 0..7 -> K = kk+8h+j ; elems 8..15 -> K = kk+8h+16+j
        b0.u[0] = *reinterpret_cast<const uint4*>(wrow0 + kk + 8 * h);
        b0.u[1] = *reinterpret_cast<const uint4*>(wrow0 + kk + 8 * h + 16);
        b1.u[0] = *reinterpret_cast<const uint4*>(wrow1 + kk + 8 * h);
        b1.u[1] = *reinterpret_cast<const uint4*>(wrow1 + kk + 8 * h + 16);
        c0 = __builtin_amdgcn_wmma_f32_16x16x32_bf16(false, a.v, false, b0.v,
                                                     (short)0, c0, false, false);
        c1 = __builtin_amdgcn_wmma_f32_16x16x32_bf16(false, a.v, false, b1.v,
                                                     (short)0, c1, false, false);
    }

    // C layout: lane col = ln, VGPR r -> row = r + 8h
#pragma unroll
    for (int r = 0; r < 8; ++r) {
        int m = mBase + r + 8 * h;
        C[(size_t)m * N + nBase + ln]      = c0[r];
        C[(size_t)m * N + nBase + 16 + ln] = c1[r];
    }
}

// ---------------------------------------------------------------------------
// Cosine-norm + scale. One wave per (b,n,h) row of 64 (2 elems/lane).
//   Qbf[(b*H+h)*N + n][d] = (q/||q||) * s_eff * sqrt(HD)   (logit scale folded into Q)
//   Kbf[(b*H+h)*N + n][d] = (k/||k||) * s_eff
//   Vt [(b*H+h)*HD + d][n] = v                             (transposed for PV GEMM)
// ---------------------------------------------------------------------------
__global__ void __launch_bounds__(256)
norm_scale_kernel(const float* __restrict__ qf,
                  const float* __restrict__ kf,
                  const float* __restrict__ vf,
                  const float* __restrict__ s_qk,
                  unsigned short* __restrict__ Qbf,
                  unsigned short* __restrict__ Kbf,
                  unsigned short* __restrict__ Vt) {
    const int lane = threadIdx.x & 31;
    const int wave = threadIdx.x >> 5;
    const int row  = blockIdx.x * 8 + wave;      // over B*SEQ*HEADS
    const int b    = row / (SEQ * HEADS);
    const int rem  = row % (SEQ * HEADS);
    const int n    = rem / HEADS;
    const int hh   = rem % HEADS;
    const int d0   = lane * 2;

    const size_t src = ((size_t)b * SEQ + n) * DIM + hh * HD;
    float q0 = qf[src + d0], q1 = qf[src + d0 + 1];
    float k0 = kf[src + d0], k1 = kf[src + d0 + 1];
    float v0 = vf[src + d0], v1 = vf[src + d0 + 1];

    float qs = q0 * q0 + q1 * q1;
    float ks = k0 * k0 + k1 * k1;
#pragma unroll
    for (int m = 16; m >= 1; m >>= 1) {
        qs += __shfl_xor(qs, m, 32);
        ks += __shfl_xor(ks, m, 32);
    }
    float qinv = 1.0f / fmaxf(sqrtf(qs), 1e-6f);
    float kinv = 1.0f / fmaxf(sqrtf(ks), 1e-6f);

    const float SQRT_DIM = 27.712812921102035f;  // sqrt(768) = init/scale factor
    const float LOGIT    = 8.0f;                 // sqrt(HEAD_DIM)
    float s0 = s_qk[hh * HD + d0]     * SQRT_DIM;
    float s1 = s_qk[hh * HD + d0 + 1] * SQRT_DIM;

    const size_t dstRow = ((size_t)(b * HEADS + hh) * SEQ + n) * HD;
    Qbf[dstRow + d0]     = f2bf(q0 * qinv * s0 * LOGIT);
    Qbf[dstRow + d0 + 1] = f2bf(q1 * qinv * s1 * LOGIT);
    Kbf[dstRow + d0]     = f2bf(k0 * kinv * s0);
    Kbf[dstRow + d0 + 1] = f2bf(k1 * kinv * s1);

    const size_t vtB = (size_t)(b * HEADS + hh) * HD * SEQ + n;
    Vt[vtB + (size_t)d0 * SEQ]       = f2bf(v0);
    Vt[vtB + (size_t)(d0 + 1) * SEQ] = f2bf(v1);
}

// ---------------------------------------------------------------------------
// Flash attention: one wave (32 threads) per 16-query tile per (b,h).
// Online softmax over 32-key chunks; K/V are L2-resident (~25 MB total).
// grid.x = B*HEADS*(SEQ/16) = 6144, block = 32.
// ---------------------------------------------------------------------------
__global__ void __launch_bounds__(32)
flash_attn_kernel(const unsigned short* __restrict__ Qbf,
                  const unsigned short* __restrict__ Kbf,
                  const unsigned short* __restrict__ Vt,
                  unsigned short* __restrict__ Obf) {
    __shared__ __align__(16) unsigned short Plds[16 * 32];

    const int lane = threadIdx.x & 31;
    const int h    = lane >> 4;
    const int ln   = lane & 15;
    const int qt   = blockIdx.x % (SEQ / 16);
    const int bh   = blockIdx.x / (SEQ / 16);

    const unsigned short* Qb = Qbf + ((size_t)bh * SEQ + qt * 16) * HD;
    const unsigned short* Kb = Kbf + (size_t)bh * SEQ * HD;
    const unsigned short* Vb = Vt  + (size_t)bh * HD * SEQ;

    // Q as two A-fragments (d-chunks [0,32) and [32,64))
    FragB16 qa0, qa1;
    {
        const unsigned short* qrow = Qb + (size_t)ln * HD;
        qa0.u[0] = *reinterpret_cast<const uint4*>(qrow + 16 * h);
        qa0.u[1] = *reinterpret_cast<const uint4*>(qrow + 16 * h + 8);
        qa1.u[0] = *reinterpret_cast<const uint4*>(qrow + 32 + 16 * h);
        qa1.u[1] = *reinterpret_cast<const uint4*>(qrow + 32 + 16 * h + 8);
    }

    v8f o0 = {}, o1 = {}, o2 = {}, o3 = {};
    float mrow[8], lrow[8];
#pragma unroll
    for (int r = 0; r < 8; ++r) { mrow[r] = -1e30f; lrow[r] = 0.0f; }

    for (int jt = 0; jt < SEQ; jt += 32) {
        // ---- S = Q K^T for 32 keys: two 16x16 tiles, each 2 WMMAs over d=64
        v8f s0 = {}, s1 = {};
        {
            const unsigned short* kr0 = Kb + (size_t)(jt + ln) * HD;
            const unsigned short* kr1 = Kb + (size_t)(jt + 16 + ln) * HD;
            FragB16 b0, b1;
            b0.u[0] = *reinterpret_cast<const uint4*>(kr0 + 16 * h);
            b0.u[1] = *reinterpret_cast<const uint4*>(kr0 + 16 * h + 8);
            b1.u[0] = *reinterpret_cast<const uint4*>(kr0 + 32 + 16 * h);
            b1.u[1] = *reinterpret_cast<const uint4*>(kr0 + 32 + 16 * h + 8);
            s0 = __builtin_amdgcn_wmma_f32_16x16x32_bf16(false, qa0.v, false, b0.v,
                                                         (short)0, s0, false, false);
            s0 = __builtin_amdgcn_wmma_f32_16x16x32_bf16(false, qa1.v, false, b1.v,
                                                         (short)0, s0, false, false);
            b0.u[0] = *reinterpret_cast<const uint4*>(kr1 + 16 * h);
            b0.u[1] = *reinterpret_cast<const uint4*>(kr1 + 16 * h + 8);
            b1.u[0] = *reinterpret_cast<const uint4*>(kr1 + 32 + 16 * h);
            b1.u[1] = *reinterpret_cast<const uint4*>(kr1 + 32 + 16 * h + 8);
            s1 = __builtin_amdgcn_wmma_f32_16x16x32_bf16(false, qa0.v, false, b0.v,
                                                         (short)0, s1, false, false);
            s1 = __builtin_amdgcn_wmma_f32_16x16x32_bf16(false, qa1.v, false, b1.v,
                                                         (short)0, s1, false, false);
        }

        // ---- online softmax; row (r+8h) stats reduced within each 16-lane half
        float p0[8], p1[8];
#pragma unroll
        for (int r = 0; r < 8; ++r) {
            float v = fmaxf(s0[r], s1[r]);
            v = fmaxf(v, __shfl_xor(v, 1, 32));
            v = fmaxf(v, __shfl_xor(v, 2, 32));
            v = fmaxf(v, __shfl_xor(v, 4, 32));
            v = fmaxf(v, __shfl_xor(v, 8, 32));
            float mn    = fmaxf(mrow[r], v);
            float alpha = __expf(mrow[r] - mn);
            mrow[r] = mn;
            o0[r] *= alpha; o1[r] *= alpha; o2[r] *= alpha; o3[r] *= alpha;
            p0[r] = __expf(s0[r] - mn);
            p1[r] = __expf(s1[r] - mn);
            float ps = p0[r] + p1[r];
            ps += __shfl_xor(ps, 1, 32);
            ps += __shfl_xor(ps, 2, 32);
            ps += __shfl_xor(ps, 4, 32);
            ps += __shfl_xor(ps, 8, 32);
            lrow[r] = lrow[r] * alpha + ps;
        }

        // ---- remap P (C-layout) -> A-layout through LDS
#pragma unroll
        for (int r = 0; r < 8; ++r) {
            Plds[(r + 8 * h) * 32 + ln]      = f2bf(p0[r]);
            Plds[(r + 8 * h) * 32 + 16 + ln] = f2bf(p1[r]);
        }
        __syncthreads();
        FragB16 pa;
        pa.u[0] = *reinterpret_cast<const uint4*>(&Plds[ln * 32 + 16 * h]);
        pa.u[1] = *reinterpret_cast<const uint4*>(&Plds[ln * 32 + 16 * h + 8]);
        __syncthreads();

        // ---- O += P @ V ; Vt gives contiguous B-fragment loads (key dim)
        {
            FragB16 vb;
            const unsigned short* vc = Vb + jt + 16 * h;
            vb.u[0] = *reinterpret_cast<const uint4*>(vc + (size_t)ln * SEQ);
            vb.u[1] = *reinterpret_cast<const uint4*>(vc + (size_t)ln * SEQ + 8);
            o0 = __builtin_amdgcn_wmma_f32_16x16x32_bf16(false, pa.v, false, vb.v,
                                                         (short)0, o0, false, false);
            vb.u[0] = *reinterpret_cast<const uint4*>(vc + (size_t)(16 + ln) * SEQ);
            vb.u[1] = *reinterpret_cast<const uint4*>(vc + (size_t)(16 + ln) * SEQ + 8);
            o1 = __builtin_amdgcn_wmma_f32_16x16x32_bf16(false, pa.v, false, vb.v,
                                                         (short)0, o1, false, false);
            vb.u[0] = *reinterpret_cast<const uint4*>(vc + (size_t)(32 + ln) * SEQ);
            vb.u[1] = *reinterpret_cast<const uint4*>(vc + (size_t)(32 + ln) * SEQ + 8);
            o2 = __builtin_amdgcn_wmma_f32_16x16x32_bf16(false, pa.v, false, vb.v,
                                                         (short)0, o2, false, false);
            vb.u[0] = *reinterpret_cast<const uint4*>(vc + (size_t)(48 + ln) * SEQ);
            vb.u[1] = *reinterpret_cast<const uint4*>(vc + (size_t)(48 + ln) * SEQ + 8);
            o3 = __builtin_amdgcn_wmma_f32_16x16x32_bf16(false, pa.v, false, vb.v,
                                                         (short)0, o3, false, false);
        }
    }

    // ---- epilogue: normalize, write bf16 attn output in [B, N, H*HD] layout
    const int b  = bh / HEADS;
    const int hh = bh % HEADS;
#pragma unroll
    for (int r = 0; r < 8; ++r) {
        float inv = 1.0f / lrow[r];
        size_t grow = ((size_t)b * SEQ + qt * 16 + r + 8 * h) * DIM + hh * HD;
        Obf[grow + ln]      = f2bf(o0[r] * inv);
        Obf[grow + 16 + ln] = f2bf(o1[r] * inv);
        Obf[grow + 32 + ln] = f2bf(o2[r] * inv);
        Obf[grow + 48 + ln] = f2bf(o3[r] * inv);
    }
}

// ---------------------------------------------------------------------------
// Host-side launch
// ---------------------------------------------------------------------------
extern "C" void kernel_launch(void* const* d_in, const int* in_sizes, int n_in,
                              void* d_out, int out_size, void* d_ws, size_t ws_size,
                              hipStream_t stream) {
    (void)in_sizes; (void)n_in; (void)out_size; (void)ws_size;
    const float* x    = (const float*)d_in[0];
    const float* Wq   = (const float*)d_in[1];
    const float* Wk   = (const float*)d_in[2];
    const float* Wv   = (const float*)d_in[3];
    const float* Wo   = (const float*)d_in[4];
    const float* s_qk = (const float*)d_in[5];

    char* ws = (char*)d_ws;
    size_t off = 0;
    auto alloc = [&](size_t bytes) -> void* {
        void* p = ws + off;
        off += (bytes + 255) & ~(size_t)255;
        return p;
    };

    const size_t actN = (size_t)MROWS * DIM;   // 6,291,456
    const size_t wN   = (size_t)DIM * DIM;     //   589,824

    unsigned short* xbf   = (unsigned short*)alloc(actN * 2);
    unsigned short* wqbf  = (unsigned short*)alloc(wN * 2);
    unsigned short* wkbf  = (unsigned short*)alloc(wN * 2);
    unsigned short* wvbf  = (unsigned short*)alloc(wN * 2);
    unsigned short* wobf  = (unsigned short*)alloc(wN * 2);
    float*          qf    = (float*)alloc(actN * 4);
    float*          kf    = (float*)alloc(actN * 4);
    float*          vf    = (float*)alloc(actN * 4);
    unsigned short* Qbf   = (unsigned short*)alloc(actN * 2);
    unsigned short* Kbf   = (unsigned short*)alloc(actN * 2);
    unsigned short* Vt    = (unsigned short*)alloc(actN * 2);
    unsigned short* attnb = (unsigned short*)alloc(actN * 2);

    // 1) convert activations + weights to bf16
    cvt_f32_bf16_kernel<<<4096, 256, 0, stream>>>(x,  xbf,  (int)actN);
    cvt_f32_bf16_kernel<<<1024, 256, 0, stream>>>(Wq, wqbf, (int)wN);
    cvt_f32_bf16_kernel<<<1024, 256, 0, stream>>>(Wk, wkbf, (int)wN);
    cvt_f32_bf16_kernel<<<1024, 256, 0, stream>>>(Wv, wvbf, (int)wN);
    cvt_f32_bf16_kernel<<<1024, 256, 0, stream>>>(Wo, wobf, (int)wN);

    // 2) Q/K/V projections (f32 accumulate)
    dim3 ggrid(MROWS / 64, DIM / 64);
    gemm_nt_bf16_kernel<<<ggrid, 256, 0, stream>>>(xbf, wqbf, qf, MROWS, DIM, DIM);
    gemm_nt_bf16_kernel<<<ggrid, 256, 0, stream>>>(xbf, wkbf, kf, MROWS, DIM, DIM);
    gemm_nt_bf16_kernel<<<ggrid, 256, 0, stream>>>(xbf, wvbf, vf, MROWS, DIM, DIM);

    // 3) cosine norm + scale, repack per-head (V transposed)
    norm_scale_kernel<<<(BATCH * SEQ * HEADS) / 8, 256, 0, stream>>>(
        qf, kf, vf, s_qk, Qbf, Kbf, Vt);

    // 4) flash attention
    flash_attn_kernel<<<BATCH * HEADS * (SEQ / 16), 32, 0, stream>>>(
        Qbf, Kbf, Vt, attnb);

    // 5) output projection -> f32 d_out
    gemm_nt_bf16_kernel<<<ggrid, 256, 0, stream>>>(attnb, wobf, (float*)d_out,
                                                   MROWS, DIM, DIM);
}
